// Vanilla_GNN_57234734186920
// MI455X (gfx1250) — compile-verified, matrix operand order checked by reading, not run on hardware
//
#include <hip/hip_runtime.h>
#include <hip/hip_bf16.h>

typedef __attribute__((ext_vector_type(16))) _Float16 v16h;
typedef __attribute__((ext_vector_type(8)))  float    v8f;

// ---------------------------------------------------------------------------
// Degree / normalization:  deg[i] = 1 (self loop) + #edges with dst == i
//                          dinv[i] = rsqrt(deg[i])      (deg >= 1 always)
// ---------------------------------------------------------------------------
__global__ void k_deg_init(float* __restrict__ deg, int n) {
  int i = blockIdx.x * blockDim.x + threadIdx.x;
  if (i < n) deg[i] = 1.0f;
}

__global__ void k_deg_accum(const int* __restrict__ dst, float* __restrict__ deg, int E) {
  int e = blockIdx.x * blockDim.x + threadIdx.x;
  if (e < E) atomicAdd(&deg[dst[e]], 1.0f);
}

__global__ void k_deg_rsqrt(float* __restrict__ deg, int n) {
  int i = blockIdx.x * blockDim.x + threadIdx.x;
  if (i < n) deg[i] = rsqrtf(deg[i]);
}

__global__ void k_zero(float* __restrict__ p, int n) {
  int i = blockIdx.x * blockDim.x + threadIdx.x;
  if (i < n) p[i] = 0.0f;
}

// ---------------------------------------------------------------------------
// Fragment packing into the ISA 16-bit WMMA layout.
// Within a 32-wide K step, K-offset r maps to:
//   kh = (r>>3)&1            (lane[4]: which half-wave holds it)
//   j  = ((r>>4)<<3)|(r&7)   (position within the lane's 16 halfs)
// so kh=0 holds r={0..7,16..23}, kh=1 holds r={8..15,24..31}  (ISA table).
//
// A pack (per M row):   P[(((i*(K/32)+t)*2 + kh)*16 + j] = A[i*K + 32t + r]
// B pack (per N col):   P[(((t*N + n)*2   + kh)*16 + j]  = B[(32t+r)*N + n]
// Each lane then loads its whole 16-half fragment as one contiguous 32B read.
// ---------------------------------------------------------------------------
__global__ void k_pack_a(const float* __restrict__ A, _Float16* __restrict__ P,
                         int n, int K) {
  int idx = blockIdx.x * blockDim.x + threadIdx.x;
  if (idx >= n * K) return;
  int i = idx / K, f = idx - i * K;
  int t = f >> 5, r = f & 31;
  int kh = (r >> 3) & 1;
  int j  = ((r >> 4) << 3) | (r & 7);
  P[(((size_t)i * (K >> 5) + t) * 2 + kh) * 16 + j] = (_Float16)A[idx];
}

__global__ void k_pack_b(const float* __restrict__ B, _Float16* __restrict__ P,
                         int K, int N) {
  int idx = blockIdx.x * blockDim.x + threadIdx.x;
  if (idx >= K * N) return;
  int k = idx / N, nn = idx - k * N;
  int t = k >> 5, r = k & 31;
  int kh = (r >> 3) & 1;
  int j  = ((r >> 4) << 3) | (r & 7);
  P[(((size_t)t * N + nn) * 2 + kh) * 16 + j] = (_Float16)B[idx];
}

// ---------------------------------------------------------------------------
// C[M,N] = A[M,K] @ B[K,N] from pre-packed f16 fragments, f32 accumulate.
// One wave32 per 16x16 tile, 4 waves/block.  K templated -> fully unrolled
// chain of K/32 v_wmma_f32_16x16x32_f16 with 2 vector loads each.
// ---------------------------------------------------------------------------
template <int K>
__global__ void k_gemm_wmma(const _Float16* __restrict__ PA,
                            const _Float16* __restrict__ PB,
                            float* __restrict__ C, int M, int N) {
  const int lane   = threadIdx.x & 31;
  const int wave   = threadIdx.x >> 5;
  const int tilesN = N >> 4;
  const int tile   = blockIdx.x * (blockDim.x >> 5) + wave;
  const int tm     = tile / tilesN;
  const int tn     = tile - tm * tilesN;
  if (tm * 16 >= M) return;                         // wave-uniform exit

  const int mn = lane & 15;                         // A row / B col of this lane
  const int kh = lane >> 4;                         // K-half selector
  constexpr int T = K >> 5;                         // 32-K steps

  int arow = tm * 16 + mn;
  if (arow >= M) arow = M - 1;                      // safe clamp for ragged M
  const _Float16* __restrict__ pa = PA + (((size_t)arow * T) * 2 + kh) * 16;
  const _Float16* __restrict__ pb = PB + ((size_t)(tn * 16 + mn) * 2 + kh) * 16;

  v8f acc = {};
#pragma unroll
  for (int t = 0; t < T; ++t) {
    const v16h av = *(const v16h*)(pa + t * 32);              // 32B contiguous
    const v16h bv = *(const v16h*)(pb + (size_t)t * N * 32);  // 32B contiguous
    acc = __builtin_amdgcn_wmma_f32_16x16x32_f16(false, av, false, bv,
                                                 (short)0, acc, false, false);
  }

  const int m0 = tm * 16 + kh * 8;
  float* __restrict__ crow = C + (size_t)m0 * N + tn * 16 + mn;
  if (tm * 16 + 16 <= M) {                          // uniform: full tile, no guards
#pragma unroll
    for (int r = 0; r < 8; ++r) crow[(size_t)r * N] = acc[r];
  } else {
#pragma unroll
    for (int r = 0; r < 8; ++r)
      if (m0 + r < M) crow[(size_t)r * N] = acc[r];
  }
}

// ---------------------------------------------------------------------------
// Edge scatter: agg[dst] += dinv[src]*dinv[dst] * h[src].  One wave per edge,
// lane-parallel features; f32 atomic-adds resolve in L2 (all arrays L2-resident).
// ---------------------------------------------------------------------------
__global__ void k_scatter_d128(const int* __restrict__ src, const int* __restrict__ dst,
                               const float* __restrict__ dinv, const float* __restrict__ h,
                               float* __restrict__ agg, int E) {
  const int lane = threadIdx.x & 31;
  const int e    = blockIdx.x * (blockDim.x >> 5) + (threadIdx.x >> 5);
  if (e >= E) return;
  const int   s = src[e], d = dst[e];
  const float w = dinv[s] * dinv[d];
  const float4 v = *(const float4*)(h + (size_t)s * 128 + lane * 4);
  float* __restrict__ ap = agg + (size_t)d * 128 + lane * 4;
  atomicAdd(ap + 0, w * v.x);
  atomicAdd(ap + 1, w * v.y);
  atomicAdd(ap + 2, w * v.z);
  atomicAdd(ap + 3, w * v.w);
}

__global__ void k_scatter_d64(const int* __restrict__ src, const int* __restrict__ dst,
                              const float* __restrict__ dinv, const float* __restrict__ h,
                              float* __restrict__ agg, int E) {
  const int lane = threadIdx.x & 31;
  const int e    = blockIdx.x * (blockDim.x >> 5) + (threadIdx.x >> 5);
  if (e >= E) return;
  const int   s = src[e], d = dst[e];
  const float w = dinv[s] * dinv[d];
  const float2 v = *(const float2*)(h + (size_t)s * 64 + lane * 2);
  float* __restrict__ ap = agg + (size_t)d * 64 + lane * 2;
  atomicAdd(ap + 0, w * v.x);
  atomicAdd(ap + 1, w * v.y);
}

// ---------------------------------------------------------------------------
// Finalize: fold in self-loop term dinv[i]^2 * h[i] + bias (+ ReLU layer 1)
// ---------------------------------------------------------------------------
__global__ void k_finalize_relu(float* __restrict__ agg, const float* __restrict__ h,
                                const float* __restrict__ dinv, const float* __restrict__ b,
                                int n, int D) {
  const int idx = blockIdx.x * blockDim.x + threadIdx.x;
  if (idx >= n * D) return;
  const int   i  = idx / D;
  const int   f  = idx - i * D;
  const float di = dinv[i];
  agg[idx] = fmaxf(agg[idx] + di * di * h[idx] + b[f], 0.0f);
}

__global__ void k_finalize_out(float* __restrict__ out, const float* __restrict__ h,
                               const float* __restrict__ dinv, const float* __restrict__ b,
                               int n, int D) {
  const int idx = blockIdx.x * blockDim.x + threadIdx.x;
  if (idx >= n * D) return;
  const int   i  = idx / D;
  const int   f  = idx - i * D;
  const float di = dinv[i];
  out[idx] = out[idx] + di * di * h[idx] + b[f];
}

// ---------------------------------------------------------------------------
extern "C" void kernel_launch(void* const* d_in, const int* in_sizes, int n_in,
                              void* d_out, int out_size, void* d_ws, size_t ws_size,
                              hipStream_t stream) {
  const float* x  = (const float*)d_in[0];
  const int*   ei = (const int*)  d_in[1];
  const float* W1 = (const float*)d_in[2];
  const float* b1 = (const float*)d_in[3];
  const float* W2 = (const float*)d_in[4];
  const float* b2 = (const float*)d_in[5];
  float*       out = (float*)d_out;

  const int Din  = 128;
  const int Dh   = in_sizes[3];          // 128
  const int Dout = in_sizes[5];          // 64
  const int n    = in_sizes[0] / Din;    // 10000
  const int E    = in_sizes[1] / 2;      // 640000
  const int* src = ei;                   // edge_index[0]
  const int* dst = ei + E;               // edge_index[1]

  // --- workspace carve-out (256B aligned blocks), ~15.5 MB total ---
  char*  ws  = (char*)d_ws;
  size_t off = 0;
  auto alloc = [&](size_t bytes) -> void* {
    void* p = ws + off;
    off = (off + bytes + 255) & ~(size_t)255;
    return p;
  };
  float*     dinv = (float*)    alloc((size_t)n * 4);           // deg -> dinv
  float*     h1   = (float*)    alloc((size_t)n * Dh * 4);
  float*     agg1 = (float*)    alloc((size_t)n * Dh * 4);      // agg1 -> z1 in place
  float*     h2   = (float*)    alloc((size_t)n * Dout * 4);
  _Float16*  pA   = (_Float16*) alloc((size_t)n * Din * 2);     // reused for z1 pack
  _Float16*  pB1  = (_Float16*) alloc((size_t)Din * Dh * 2);
  _Float16*  pB2  = (_Float16*) alloc((size_t)Dh * Dout * 2);

  const int T = 256;

  // --- gcn_norm ---
  k_deg_init <<<(n + T - 1) / T, T, 0, stream>>>(dinv, n);
  k_deg_accum<<<(E + T - 1) / T, T, 0, stream>>>(dst, dinv, E);
  k_deg_rsqrt<<<(n + T - 1) / T, T, 0, stream>>>(dinv, n);

  // --- zero accumulators (graph replayed: must re-zero every call) ---
  k_zero<<<(n * Dh   + T - 1) / T, T, 0, stream>>>(agg1, n * Dh);
  k_zero<<<(n * Dout + T - 1) / T, T, 0, stream>>>(out,  n * Dout);

  // --- pack weights (tiny) and x into WMMA fragment layout ---
  k_pack_b<<<(Din * Dh   + T - 1) / T, T, 0, stream>>>(W1, pB1, Din, Dh);
  k_pack_b<<<(Dh  * Dout + T - 1) / T, T, 0, stream>>>(W2, pB2, Dh, Dout);
  k_pack_a<<<(n * Din + T - 1) / T, T, 0, stream>>>(x, pA, n, Din);

  // --- layer 1: h1 = x @ W1 ; scatter ; relu(+self loop, +b1) ---
  const int tiles1 = ((n + 15) / 16) * (Dh / 16);
  k_gemm_wmma<128><<<(tiles1 + 3) / 4, 128, 0, stream>>>(pA, pB1, h1, n, Dh);
  k_scatter_d128<<<(E + 7) / 8, 256, 0, stream>>>(src, dst, dinv, h1, agg1, E);
  k_finalize_relu<<<(n * Dh + T - 1) / T, T, 0, stream>>>(agg1, h1, dinv, b1, n, Dh);

  // --- layer 2: h2 = z1 @ W2 ; scatter into out ; +self loop, +b2 ---
  k_pack_a<<<(n * Dh + T - 1) / T, T, 0, stream>>>(agg1, pA, n, Dh);
  const int tiles2 = ((n + 15) / 16) * (Dout / 16);
  k_gemm_wmma<128><<<(tiles2 + 3) / 4, 128, 0, stream>>>(pA, pB2, h2, n, Dout);
  k_scatter_d64<<<(E + 7) / 8, 256, 0, stream>>>(src, dst, dinv, h2, out, E);
  k_finalize_out<<<(n * Dout + T - 1) / T, T, 0, stream>>>(out, h2, dinv, b2, n, Dout);
}
